// RNN_45973329937260
// MI455X (gfx1250) — matrix operand order repaired and checked by reading, not compiled, not run on hardware
//
#include <hip/hip_runtime.h>
#include <hip/hip_bf16.h>
#include <math.h>

// ---------------- problem sizes ----------------
#define SEQ   1024
#define BATCH 64
#define INP   512
#define HD    1024
#define HFC   512
#define NCHUNK 32            // workgroups in the recurrence (32 cols each)
#define COLS  (HD / NCHUNK)  // 32 output columns per workgroup

typedef __bf16 bf16;
typedef __attribute__((ext_vector_type(16))) __bf16 bf16x16;
typedef __attribute__((ext_vector_type(8)))  float  floatx8;
typedef __attribute__((ext_vector_type(4)))  unsigned int uint32x4;

// ---------------- fast activations (CDNA5 V_TANH_F32 when available) --------
__device__ __forceinline__ float fast_tanh(float x) {
#if __has_builtin(__builtin_amdgcn_tanhf)
  return __builtin_amdgcn_tanhf(x);
#elif __has_builtin(__builtin_amdgcn_tanh_f32)
  return __builtin_amdgcn_tanh_f32(x);
#else
  // branch-free: tanh(x) = 1 - 2/(1 + exp(2x)); __expf lowers to v_exp_f32
  float e = __expf(2.0f * x);
  return 1.0f - 2.0f / (e + 1.0f);
#endif
}
__device__ __forceinline__ float fast_sigmoid(float x) {
  return 1.0f / (1.0f + __expf(-x));
}

// ---------------- WMMA fragment helpers (CDNA5 ISA 7.12.2 layouts) ----------
// A fragment: 16x32 bf16, row-major source with leading dim `ld`.
// lanes 0-15: M=lane, K groups {0-7,16-23}; lanes 16-31: M=lane-16, K {8-15,24-31}
__device__ __forceinline__ bf16x16 loadA_bf16(const bf16* base, int ld) {
  const int lane = threadIdx.x & 31;
  const int m = lane & 15, half = lane >> 4;
  const bf16* p = base + (size_t)m * ld + half * 8;
  union { uint32x4 q[2]; bf16x16 f; } u;
  u.q[0] = *reinterpret_cast<const uint32x4*>(p);        // K = half*8 .. +7
  u.q[1] = *reinterpret_cast<const uint32x4*>(p + 16);   // K = 16+half*8 .. +7
  return u.f;
}

// A fragment from fp32 source, converting (and optionally scaling) to bf16.
__device__ __forceinline__ bf16x16 loadA_f32(const float* base, int ld, float scale) {
  const int lane = threadIdx.x & 31;
  const int m = lane & 15, half = lane >> 4;
  const float* p = base + (size_t)m * ld + half * 8;
  bf16x16 f;
#pragma unroll
  for (int i = 0; i < 8; ++i) {
    f[i]     = (bf16)(p[i] * scale);
    f[i + 8] = (bf16)(p[16 + i] * scale);
  }
  return f;
}

// B fragment: 32x16 bf16 (KxN). Source stored as W[n][k] row-major (ld = K dim).
// lanes 0-15: N=lane, K {0-7,16-23}; lanes 16-31: N=lane-16, K {8-15,24-31}
__device__ __forceinline__ bf16x16 loadB_bf16(const bf16* base, int ld) {
  const int lane = threadIdx.x & 31;
  const int n = lane & 15, kh = lane >> 4;
  const bf16* p = base + (size_t)n * ld + kh * 8;
  union { uint32x4 q[2]; bf16x16 f; } u;
  u.q[0] = *reinterpret_cast<const uint32x4*>(p);
  u.q[1] = *reinterpret_cast<const uint32x4*>(p + 16);
  return u.f;
}

__device__ __forceinline__ floatx8 wmma_bf16(bf16x16 a, bf16x16 b, floatx8 c) {
  return __builtin_amdgcn_wmma_f32_16x16x32_bf16(false, a, false, b, (short)0, c,
                                                 false, false);
}

// ---------------- utility kernels ----------------
__global__ void k_cvt(const float* __restrict__ src, bf16* __restrict__ dst, int n) {
  int i = blockIdx.x * 256 + threadIdx.x;
  if (i < n) dst[i] = (bf16)src[i];
}
__global__ void k_addb(const float* a, const float* b, float* o, int n) {
  int i = blockIdx.x * 256 + threadIdx.x;
  if (i < n) o[i] = a[i] + b[i];
}
__global__ void k_zero_f(float* p, int n) {
  int i = blockIdx.x * 256 + threadIdx.x;
  if (i < n) p[i] = 0.0f;
}
__global__ void k_zero_u(unsigned* p, int n) {
  int i = blockIdx.x * 256 + threadIdx.x;
  if (i < n) p[i] = 0u;
}

// ---------------- input projection GEMM: Y = A @ W^T + bias (bf16 out) -------
// A: MxK (fp32 if AF32 else bf16), W: HD x K row-major bf16, Y: M x HD bf16.
// Block tile 64(M)x64(N), 8 waves, each wave one 16x32 strip (2 WMMA tiles).
template <bool AF32, int K>
__global__ __launch_bounds__(256) void k_proj(const void* __restrict__ Av,
                                              const bf16* __restrict__ W,
                                              const float* __restrict__ bias,
                                              bf16* __restrict__ Y) {
  const int m0 = blockIdx.y * 64;
  const int n0 = blockIdx.x * 64;
  const int w = threadIdx.x >> 5, lane = threadIdx.x & 31;
  const int mw = m0 + (w & 3) * 16;
  const int nw = n0 + (w >> 2) * 32;
  floatx8 c0 = {}, c1 = {};
  for (int k = 0; k < K; k += 32) {
    bf16x16 a;
    if (AF32) a = loadA_f32((const float*)Av + (size_t)mw * K + k, K, 1.0f);
    else      a = loadA_bf16((const bf16*)Av + (size_t)mw * K + k, K);
    bf16x16 b0 = loadB_bf16(W + (size_t)nw * K + k, K);
    bf16x16 b1 = loadB_bf16(W + (size_t)(nw + 16) * K + k, K);
    c0 = wmma_bf16(a, b0, c0);
    c1 = wmma_bf16(a, b1, c1);
  }
  const int half = lane >> 4, nl = lane & 15;
  const float bv0 = bias[nw + nl];
  const float bv1 = bias[nw + 16 + nl];
#pragma unroll
  for (int r = 0; r < 8; ++r) {
    int gm = mw + r + 8 * half;
    Y[(size_t)gm * HD + nw + nl]      = (bf16)(c0[r] + bv0);
    Y[(size_t)gm * HD + nw + 16 + nl] = (bf16)(c1[r] + bv1);
  }
}

// ---------------- recurrence: h_t = tanh(xw_t + h_{t-1} @ W^T) ---------------
// 32 persistent workgroups; each owns 32 columns of W (64KB bf16 in LDS).
// h (64x1024 bf16) ping-pongs through L2; one grid barrier per step.
// Two independent WMMA accumulation chains per wave hide the D->C latency.
__global__ __launch_bounds__(256) void k_recur(const bf16* __restrict__ W,
                                               const bf16* __restrict__ xw,
                                               bf16* __restrict__ hbuf,
                                               bf16* __restrict__ rout,
                                               float* __restrict__ pooled,
                                               unsigned* __restrict__ cnt,
                                               int layer1) {
  __shared__ bf16 Wsh[COLS * HD];  // 32 x 1024 bf16 = 64 KB
  const int n_base = blockIdx.x * COLS;
  {
    const uint32x4* src = reinterpret_cast<const uint32x4*>(W + (size_t)n_base * HD);
    uint32x4* dst = reinterpret_cast<uint32x4*>(Wsh);
    const int total = COLS * HD * (int)sizeof(bf16) / 16;  // 4096
    for (int i = threadIdx.x; i < total; i += 256) dst[i] = src[i];
  }
  __syncthreads();

  const int lane = threadIdx.x & 31;
  const int w = threadIdx.x >> 5;
  const int mt = (w & 3) * 16;          // batch-row tile base (0..48)
  const int ntl = (w >> 2) * 16;        // local column tile base (0/16)
  const int half = lane >> 4, nl = lane & 15;
  const int ng = n_base + ntl + nl;     // this lane's global output column
  const unsigned nb = gridDim.x;

  for (int t = 0; t < SEQ; ++t) {
    const bf16* hcur = hbuf + (size_t)(t & 1) * BATCH * HD;
    bf16* hnext = hbuf + (size_t)((t + 1) & 1) * BATCH * HD;
    floatx8 c0 = {}, c1 = {};
#pragma unroll 2
    for (int k = 0; k < HD; k += 64) {
      bf16x16 a0 = loadA_bf16(hcur + (size_t)mt * HD + k, HD);
      bf16x16 b0 = loadB_bf16(Wsh + ntl * HD + k, HD);
      bf16x16 a1 = loadA_bf16(hcur + (size_t)mt * HD + k + 32, HD);
      bf16x16 b1 = loadB_bf16(Wsh + ntl * HD + k + 32, HD);
      c0 = wmma_bf16(a0, b0, c0);     // two independent accumulation chains
      c1 = wmma_bf16(a1, b1, c1);
    }
    const bf16* xwt = xw + (size_t)t * BATCH * HD;
    float psum = 0.0f;
#pragma unroll
    for (int r = 0; r < 8; ++r) {
      int m = mt + r + 8 * half;
      float v = (c0[r] + c1[r]) + (float)xwt[(size_t)m * HD + ng];
      v = fast_tanh(v);
      hnext[(size_t)m * HD + ng] = (bf16)v;
      if (rout) rout[((size_t)t * BATCH + m) * HD + ng] = (bf16)v;
      psum += v;
    }
    if (layer1) {  // fused mean-over-batch pooling (sum; 1/64 folded into FC1)
      psum += __shfl_down(psum, 16, 32);
      if (lane < 16) atomicAdd(&pooled[(size_t)t * HD + ng], psum);
    }
    // ---- one grid barrier per step (monotonic counter, sense via target) ----
    __threadfence();
    __syncthreads();
    if (threadIdx.x == 0) {
      __hip_atomic_fetch_add(cnt, 1u, __ATOMIC_RELEASE, __HIP_MEMORY_SCOPE_AGENT);
      const unsigned target = nb * (unsigned)(t + 1);
      while (__hip_atomic_load(cnt, __ATOMIC_ACQUIRE, __HIP_MEMORY_SCOPE_AGENT) <
             target) {
        __builtin_amdgcn_s_sleep(1);
      }
    }
    __syncthreads();
  }
}

// ---------------- FC1: h1 = relu(pooled/64 @ W1^T + b1) ---------------------
__global__ __launch_bounds__(256) void k_fc1(const float* __restrict__ pooled,
                                             const bf16* __restrict__ W1,
                                             const float* __restrict__ b1,
                                             float* __restrict__ outp) {
  const int m0 = blockIdx.y * 64;
  const int n0 = blockIdx.x * 64;
  const int w = threadIdx.x >> 5, lane = threadIdx.x & 31;
  const int mw = m0 + (w & 3) * 16;
  const int nw = n0 + (w >> 2) * 32;
  floatx8 c0 = {}, c1 = {};
  const float inv_b = 1.0f / (float)BATCH;
  for (int k = 0; k < HD; k += 32) {
    bf16x16 a = loadA_f32(pooled + (size_t)mw * HD + k, HD, inv_b);
    bf16x16 b0 = loadB_bf16(W1 + (size_t)nw * HD + k, HD);
    bf16x16 b1f = loadB_bf16(W1 + (size_t)(nw + 16) * HD + k, HD);
    c0 = wmma_bf16(a, b0, c0);
    c1 = wmma_bf16(a, b1f, c1);
  }
  const int half = lane >> 4, nl = lane & 15;
  const float bv0 = b1[nw + nl];
  const float bv1 = b1[nw + 16 + nl];
#pragma unroll
  for (int r = 0; r < 8; ++r) {
    int gm = mw + r + 8 * half;
    outp[(size_t)gm * HFC + nw + nl]      = fmaxf(c0[r] + bv0, 0.0f);
    outp[(size_t)gm * HFC + nw + 16 + nl] = fmaxf(c1[r] + bv1, 0.0f);
  }
}

// ---------------- FC2: out[s] = sigmoid(h1[s] . W2 + b2) --------------------
__global__ __launch_bounds__(128) void k_fc2(const float* __restrict__ h1,
                                             const float* __restrict__ W2,
                                             const float* __restrict__ b2,
                                             float* __restrict__ out) {
  __shared__ float red[128];
  const int s = blockIdx.x;
  float acc = 0.0f;
  for (int i = threadIdx.x; i < HFC; i += 128)
    acc += h1[(size_t)s * HFC + i] * W2[i];
  red[threadIdx.x] = acc;
  __syncthreads();
  for (int off = 64; off > 0; off >>= 1) {
    if (threadIdx.x < off) red[threadIdx.x] += red[threadIdx.x + off];
    __syncthreads();
  }
  if (threadIdx.x == 0) out[s] = fast_sigmoid(red[0] + b2[0]);
}

// ---------------- workspace layout (bytes, all 256B aligned) ----------------
static constexpr size_t XW_BYTES  = (size_t)SEQ * BATCH * HD * 2;   // 128 MB
static constexpr size_t R0_BYTES  = XW_BYTES;                       // 128 MB
static constexpr size_t WIH0_B = (size_t)HD * INP * 2;              // 1 MB
static constexpr size_t WHH_B  = (size_t)HD * HD * 2;               // 2 MB
static constexpr size_t W1_B   = (size_t)HFC * HD * 2;              // 1 MB
static constexpr size_t OFF_XW   = 0;
static constexpr size_t OFF_R0   = OFF_XW + XW_BYTES;
static constexpr size_t OFF_WIH0 = OFF_R0 + R0_BYTES;
static constexpr size_t OFF_WHH0 = OFF_WIH0 + WIH0_B;
static constexpr size_t OFF_WIH1 = OFF_WHH0 + WHH_B;
static constexpr size_t OFF_WHH1 = OFF_WIH1 + WHH_B;
static constexpr size_t OFF_W1   = OFF_WHH1 + WHH_B;
static constexpr size_t OFF_B0   = OFF_W1 + W1_B;
static constexpr size_t OFF_B1   = OFF_B0 + HD * 4;
static constexpr size_t OFF_HBUF = OFF_B1 + HD * 4;                 // 2x64x1024 bf16
static constexpr size_t OFF_POOL = OFF_HBUF + (size_t)2 * BATCH * HD * 2;
static constexpr size_t OFF_FC1  = OFF_POOL + (size_t)SEQ * HD * 4;
static constexpr size_t OFF_CNT  = OFF_FC1 + (size_t)SEQ * HFC * 4;

extern "C" void kernel_launch(void* const* d_in, const int* in_sizes, int n_in,
                              void* d_out, int out_size, void* d_ws, size_t ws_size,
                              hipStream_t stream) {
  const float* x      = (const float*)d_in[0];
  const float* hidden = (const float*)d_in[1];
  const float* W_ih0  = (const float*)d_in[2];
  const float* W_hh0  = (const float*)d_in[3];
  const float* b_ih0  = (const float*)d_in[4];
  const float* b_hh0  = (const float*)d_in[5];
  const float* W_ih1  = (const float*)d_in[6];
  const float* W_hh1  = (const float*)d_in[7];
  const float* b_ih1  = (const float*)d_in[8];
  const float* b_hh1  = (const float*)d_in[9];
  const float* W1     = (const float*)d_in[10];
  const float* b1     = (const float*)d_in[11];
  const float* W2     = (const float*)d_in[12];
  const float* b2     = (const float*)d_in[13];
  float* out = (float*)d_out;

  char* ws = (char*)d_ws;
  bf16*  xwbuf = (bf16*)(ws + OFF_XW);
  bf16*  r0buf = (bf16*)(ws + OFF_R0);
  bf16*  wih0b = (bf16*)(ws + OFF_WIH0);
  bf16*  whh0b = (bf16*)(ws + OFF_WHH0);
  bf16*  wih1b = (bf16*)(ws + OFF_WIH1);
  bf16*  whh1b = (bf16*)(ws + OFF_WHH1);
  bf16*  w1b   = (bf16*)(ws + OFF_W1);
  float* bias0 = (float*)(ws + OFF_B0);
  float* bias1 = (float*)(ws + OFF_B1);
  bf16*  hbuf  = (bf16*)(ws + OFF_HBUF);
  float* pooled = (float*)(ws + OFF_POOL);
  float* fc1buf = (float*)(ws + OFF_FC1);
  unsigned* cnt = (unsigned*)(ws + OFF_CNT);

  auto blk = [](int n) { return (n + 255) / 256; };

  // ---- prep: weight conversion, bias combine, zeroing (re-done every call) --
  k_cvt<<<blk(HD * INP), 256, 0, stream>>>(W_ih0, wih0b, HD * INP);
  k_cvt<<<blk(HD * HD),  256, 0, stream>>>(W_hh0, whh0b, HD * HD);
  k_cvt<<<blk(HD * HD),  256, 0, stream>>>(W_ih1, wih1b, HD * HD);
  k_cvt<<<blk(HD * HD),  256, 0, stream>>>(W_hh1, whh1b, HD * HD);
  k_cvt<<<blk(HFC * HD), 256, 0, stream>>>(W1, w1b, HFC * HD);
  k_addb<<<blk(HD), 256, 0, stream>>>(b_ih0, b_hh0, bias0, HD);
  k_addb<<<blk(HD), 256, 0, stream>>>(b_ih1, b_hh1, bias1, HD);
  k_zero_f<<<blk(SEQ * HD), 256, 0, stream>>>(pooled, SEQ * HD);
  k_zero_u<<<1, 32, 0, stream>>>(cnt, 2);

  // ---- layer 0 --------------------------------------------------------------
  k_cvt<<<blk(BATCH * HD), 256, 0, stream>>>(hidden, hbuf, BATCH * HD);
  k_proj<true, INP><<<dim3(HD / 64, (SEQ * BATCH) / 64), 256, 0, stream>>>(
      (const void*)x, wih0b, bias0, xwbuf);
  k_recur<<<NCHUNK, 256, 0, stream>>>(whh0b, xwbuf, hbuf, r0buf, nullptr,
                                      cnt + 0, 0);

  // ---- layer 1 (xW buffer reused; pooling fused into recurrence) ------------
  k_proj<false, HD><<<dim3(HD / 64, (SEQ * BATCH) / 64), 256, 0, stream>>>(
      (const void*)r0buf, wih1b, bias1, xwbuf);
  k_cvt<<<blk(BATCH * HD), 256, 0, stream>>>(hidden + (size_t)BATCH * HD, hbuf,
                                             BATCH * HD);
  k_recur<<<NCHUNK, 256, 0, stream>>>(whh1b, xwbuf, hbuf, nullptr, pooled,
                                      cnt + 1, 1);

  // ---- FC head --------------------------------------------------------------
  k_fc1<<<dim3(HFC / 64, SEQ / 64), 256, 0, stream>>>(pooled, w1b, b1, fc1buf);
  k_fc2<<<SEQ, 128, 0, stream>>>(fc1buf, W2, b2, out);
  (void)in_sizes; (void)n_in; (void)out_size; (void)ws_size;
}
